// SymmetryAwareTaylorApproximatedAttention_3599182594292
// MI455X (gfx1250) — matrix operand order repaired and checked by reading, not compiled, not run on hardware
//
#include <hip/hip_runtime.h>
#include <hip/hip_bf16.h>

typedef __attribute__((ext_vector_type(16))) _Float16 v16h;
typedef __attribute__((ext_vector_type(8)))  float    v8f;

#define TQ    32      // queries per wave = two 16-row WMMA tiles
#define WAVES 8
#define QBLK  (TQ * WAVES)   // 256 queries per workgroup
#define KBLK  32      // keys per LDS block (two 16-key WMMA tiles)
#define DK    32      // d_key
#define DV    64      // d_val
#define KSTR  36      // padded f16 row stride for K LDS tile (bank-conflict-free)
#define VSTR  36      // padded f16 row stride for transposed V LDS tile

// Degree-3 truncated exp: 1 + x + x^2/2 + x^3/6 (Horner form matching reference)
__device__ __forceinline__ float taylor4(float x) {
    float t = __builtin_fmaf(x, 0.3333333333333333f, 1.0f); // 1 + x/3
    t = __builtin_fmaf(x * 0.5f, t, 1.0f);                  // 1 + (x/2)*t
    t = __builtin_fmaf(x, t, 1.0f);                         // 1 + x*t
    return t;
}

// Cooperative f32->f16 staging of one 32-key K/V block into (double-buffered) LDS.
__device__ __forceinline__ void stage_block(const float* __restrict__ Kg,
                                            const float* __restrict__ Vg,
                                            size_t rowbase, int j0, int tid,
                                            _Float16* kb, _Float16* vb) {
    const int key = tid >> 3;
    const int f0  = (tid & 7) * 4;
    const int c0  = (tid & 7) * 8;
    const float* krow = Kg + (rowbase + j0 + key) * DK;
    const float* vrow = Vg + (rowbase + j0 + key) * DV;

    float4 kq = *(const float4*)(krow + f0);
    union { _Float16 hx[2]; unsigned u; } p0, p1;
    p0.hx[0] = (_Float16)kq.x; p0.hx[1] = (_Float16)kq.y;
    p1.hx[0] = (_Float16)kq.z; p1.hx[1] = (_Float16)kq.w;
    *(unsigned*)(kb + key * KSTR + f0)     = p0.u;   // packed b32 DS stores
    *(unsigned*)(kb + key * KSTR + f0 + 2) = p1.u;

    float4 va  = *(const float4*)(vrow + c0);
    float4 vb4 = *(const float4*)(vrow + c0 + 4);
    vb[(c0 + 0) * VSTR + key] = (_Float16)va.x;      // transposed [dval][key]
    vb[(c0 + 1) * VSTR + key] = (_Float16)va.y;
    vb[(c0 + 2) * VSTR + key] = (_Float16)va.z;
    vb[(c0 + 3) * VSTR + key] = (_Float16)va.w;
    vb[(c0 + 4) * VSTR + key] = (_Float16)vb4.x;
    vb[(c0 + 5) * VSTR + key] = (_Float16)vb4.y;
    vb[(c0 + 6) * VSTR + key] = (_Float16)vb4.z;
    vb[(c0 + 7) * VSTR + key] = (_Float16)vb4.w;
}

__global__ __launch_bounds__(QBLK)
void taylor_attn_wmma_kernel(const float* __restrict__ Qg,
                             const float* __restrict__ Kg,
                             const float* __restrict__ Vg,
                             float* __restrict__ Og,
                             int T, int nqblk) {
    __shared__ _Float16 kbuf[2][KBLK * KSTR];  // [buf][key][feat]   f16
    __shared__ _Float16 vbuf[2][DV * VSTR];    // [buf][dval][key]   f16 (transposed)

    const int tid  = threadIdx.x;
    const int lane = tid & 31;
    const int n    = lane & 15;   // matrix column index within 16
    const int h    = lane >> 4;   // half-wave select
    // Scalar wave id -> causal tests become s_cmp/s_cbranch, EXEC never touched.
    const int wave = __builtin_amdgcn_readfirstlane(tid >> 5);

    const int bh   = blockIdx.x / nqblk;
    const int qblk = blockIdx.x % nqblk;
    const int q0wg = qblk * QBLK;          // first query of workgroup
    const int q0w  = q0wg + wave * TQ;     // first query of this wave (mult of 32)

    const float scale = 0.17677669529663687f; // 1/sqrt(32)
    const size_t rowbase = (size_t)bh * T;

    // ---- Q^T B-operands (one per 16-query tile), in registers all kernel ----
    // B operand, this code's contiguous k order (same as K A-operand): lane n = column (query), VGPR v holds the
    // f16 pair at K-dim (feature) = 2v + 16h.
    v16h qb[2];
    #pragma unroll
    for (int qt = 0; qt < 2; ++qt) {
        const float* qrow = Qg + (rowbase + q0w + 16 * qt + n) * DK;
        #pragma unroll
        for (int v = 0; v < 8; ++v) {
            float2 f = *(const float2*)(qrow + 2 * v + 16 * h);
            qb[qt][2 * v + 0] = (_Float16)(f.x * scale);
            qb[qt][2 * v + 1] = (_Float16)(f.y * scale);
        }
    }

    v8f  acc[2][4] = {{v8f{}, v8f{}, v8f{}, v8f{}}, {v8f{}, v8f{}, v8f{}, v8f{}}};
    float zacc[2] = {0.0f, 0.0f};

    const int nkb = (q0wg + QBLK) / KBLK;  // causal key-block count for this WG

    // ---- software pipeline: stage block 0, then overlap stage(j+1)/compute(j)
    stage_block(Kg, Vg, rowbase, 0, tid, kbuf[0], vbuf[0]);
    __syncthreads();

    for (int jb = 0; jb < nkb; ++jb) {
        const int j0  = jb * KBLK;
        const int cur = jb & 1;

        if (jb + 1 < nkb)
            stage_block(Kg, Vg, rowbase, j0 + KBLK, tid, kbuf[cur ^ 1], vbuf[cur ^ 1]);
        if (jb + 2 < nkb) {  // gfx1250 global_prefetch_b8 for block j+2
            const int key = tid >> 3;
            __builtin_prefetch(Kg + (rowbase + j0 + 2 * KBLK + key) * DK + (tid & 7) * 4, 0, 3);
            __builtin_prefetch(Vg + (rowbase + j0 + 2 * KBLK + key) * DV + (tid & 7) * 8, 0, 3);
        }

        // Scalar causal skip; q0w and j0 are multiples of 32, so only the
        // j0 == q0w block is the (masked) diagonal.
        if (j0 <= q0w) {
            const bool diag = (j0 == q0w);
            const _Float16* kbq = kbuf[cur];
            const _Float16* vbq = vbuf[cur];

            // ---- K-tile A-operands (16 keys x 32 feats), f16, from LDS ----
            // A layout: lane row = key (n), VGPR i holds f16 pair at feature
            // f0 = 2i + 16h.
            union { v16h v; unsigned u[8]; } ka0, ka1;
            #pragma unroll
            for (int i = 0; i < 8; ++i) {
                const int f0 = 2 * i + 16 * h;
                ka0.u[i] = *(const unsigned*)(kbq + (n)      * KSTR + f0);
                ka1.u[i] = *(const unsigned*)(kbq + (16 + n) * KSTR + f0);
            }

            // ---- S^T tiles = K_tile x Q^T, poly+mask+Z, pack to A-operand ----
            v16h sa[2];
            #pragma unroll
            for (int qt = 0; qt < 2; ++qt) {
                v8f st0 = {}, st1 = {};
                st0 = __builtin_amdgcn_wmma_f32_16x16x32_f16(false, ka0.v, false, qb[qt],
                                                             (short)0, st0, false, false);
                st1 = __builtin_amdgcn_wmma_f32_16x16x32_f16(false, ka1.v, false, qb[qt],
                                                             (short)0, st1, false, false);
                if (!diag) {            // strictly-below-diagonal: mask-free path
                    #pragma unroll
                    for (int v = 0; v < 8; ++v) {
                        float a0 = taylor4(st0[v]);
                        float a1 = taylor4(st1[v]);
                        zacc[qt] += a0 + a1;
                        sa[qt][v]     = (_Float16)a0;
                        sa[qt][8 + v] = (_Float16)a1;
                    }
                } else {                // diagonal block: triangular mask
                    const int qg = 16 * qt + n;          // query within 32-block
                    #pragma unroll
                    for (int v = 0; v < 8; ++v) {
                        const int k0 = v + 8 * h;        // key of st0[v]
                        const int k1 = k0 + 16;          // key of st1[v]
                        float a0 = (k0 <= qg) ? taylor4(st0[v]) : 0.0f;
                        float a1 = (k1 <= qg) ? taylor4(st1[v]) : 0.0f;
                        zacc[qt] += a0 + a1;
                        sa[qt][v]     = (_Float16)a0;
                        sa[qt][8 + v] = (_Float16)a1;
                    }
                }
            }

            // ---- O += S x V : 4 dval column tiles, V B-operand shared by qt ----
            #pragma unroll
            for (int t = 0; t < 4; ++t) {
                union { v16h v; unsigned u[8]; } vb;
                #pragma unroll
                for (int i = 0; i < 8; ++i) {
                    // B layout: lane = dval col (16t+n), VGPR i = key pair (i<4 ? 2i : 16+2(i-4)) + 8h
                    vb.u[i] = *(const unsigned*)(vbq + (16 * t + n) * VSTR
                                                 + ((i < 4) ? 2 * i : 16 + 2 * (i - 4)) + 8 * h);
                }
                acc[0][t] = __builtin_amdgcn_wmma_f32_16x16x32_f16(false, sa[0], false, vb.v,
                                                                   (short)0, acc[0][t],
                                                                   false, false);
                acc[1][t] = __builtin_amdgcn_wmma_f32_16x16x32_f16(false, sa[1], false, vb.v,
                                                                   (short)0, acc[1][t],
                                                                   false, false);
            }
        }
        __syncthreads();  // staged buffer ready / compute done before reuse
    }

    // ---- Z fold + normalize + store ----
    #pragma unroll
    for (int qt = 0; qt < 2; ++qt) {
        float ztot = zacc[qt] + __shfl_xor(zacc[qt], 16, 32);  // Z[lane&15] everywhere
        #pragma unroll
        for (int v = 0; v < 8; ++v) {
            const float zr = __shfl(ztot, v + 8 * h, 32);  // Z for this vgpr's row
            const float rz = 1.0f / zr;
            const int row = q0w + 16 * qt + v + 8 * h;
            float* orow = Og + (rowbase + row) * DV;
            orow[ 0 + n] = acc[qt][0][v] * rz;
            orow[16 + n] = acc[qt][1][v] * rz;
            orow[32 + n] = acc[qt][2][v] * rz;
            orow[48 + n] = acc[qt][3][v] * rz;
        }
    }
}

extern "C" void kernel_launch(void* const* d_in, const int* in_sizes, int n_in,
                              void* d_out, int out_size, void* d_ws, size_t ws_size,
                              hipStream_t stream) {
    const float* Q = (const float*)d_in[0];
    const float* K = (const float*)d_in[1];
    const float* V = (const float*)d_in[2];
    float* O = (float*)d_out;

    const int B = 4, H = 8, T = 2048;
    (void)in_sizes; (void)n_in; (void)out_size; (void)d_ws; (void)ws_size;

    const int nqblk = T / QBLK;              // 8 query blocks per head
    const int grid  = B * H * nqblk;         // 256 workgroups
    taylor_attn_wmma_kernel<<<dim3(grid), dim3(QBLK), 0, stream>>>(Q, K, V, O, T, nqblk);
}
